// PointnetSAModuleMSG_56642028700412
// MI455X (gfx1250) — matrix operand order repaired and checked by reading, not compiled, not run on hardware
//
#include <hip/hip_runtime.h>
#include <hip/hip_bf16.h>

typedef __attribute__((ext_vector_type(16))) _Float16 v16h;
typedef __attribute__((ext_vector_type(8)))  _Float16 v8h;
typedef __attribute__((ext_vector_type(8)))  float    v8f;

#define BATCH   8
#define NPTS    4096
#define CFEAT   128
#define CIN     131
#define COUT    256
#define NPOINT  1024
#define NSAMPLE 32
#define KPAD    160      // 131 padded to 5 chunks of 32 for f16 WMMA
#define R2      0.01f    // RADIUS^2
#define BN_EPS  1e-5f

// ---------------------------------------------------------------------------
// Kernel 1: convert W (256x131 f32) -> zero-padded f16 Wh (256x160)
// ---------------------------------------------------------------------------
__global__ void prep_w_kernel(const float* __restrict__ W, _Float16* __restrict__ Wh) {
    int t = blockIdx.x * 256 + threadIdx.x;
    if (t >= COUT * KPAD) return;
    int o = t / KPAD;
    int k = t - o * KPAD;
    Wh[t] = (k < CIN) ? (_Float16)W[o * CIN + k] : (_Float16)0.0f;
}

// ---------------------------------------------------------------------------
// Kernel 2: furthest point sampling. One workgroup per batch. Points live in
// registers (16 per thread). Per-step argmax: wave32 shuffle reduction (no
// barriers) + 8-wide cross-wave LDS reduction done redundantly by every
// thread -> only 2 __syncthreads per serial step (vs 9 for an LDS tree).
// Writes new_xyz (first section of d_out).
// ---------------------------------------------------------------------------
__global__ void __launch_bounds__(256)
fps_kernel(const float* __restrict__ xyz, float* __restrict__ new_xyz) {
    const int b    = blockIdx.x;
    const int tid  = threadIdx.x;
    const int wave = tid >> 5;
    const int lane = tid & 31;
    const float* xb = xyz + (size_t)b * NPTS * 3;

    float lx[16], ly[16], lz[16], md[16];
#pragma unroll
    for (int i = 0; i < 16; ++i) {
        int n = i * 256 + tid;
        lx[i] = xb[n * 3 + 0];
        ly[i] = xb[n * 3 + 1];
        lz[i] = xb[n * 3 + 2];
        md[i] = 1e10f;
    }

    __shared__ float px, py, pz;
    __shared__ float sv[8];
    __shared__ int   si[8];
    __shared__ int   sel[NPOINT];

    if (tid == 0) sel[0] = 0;
    int last = 0;

    for (int step = 1; step < NPOINT; ++step) {
        // owner of `last` broadcasts its coordinates
        if (tid == (last & 255)) {
            int slot = last >> 8;
            px = lx[slot]; py = ly[slot]; pz = lz[slot];
        }
        __syncthreads();   // (1) px visible; also fences prev-step sv/si reads
        const float bx = px, by = py, bz = pz;

        float best = -1.0f;
        int bestn = 0;
#pragma unroll
        for (int i = 0; i < 16; ++i) {
            float dx = lx[i] - bx, dy = ly[i] - by, dz = lz[i] - bz;
            float d  = dx * dx + dy * dy + dz * dz;
            float m  = fminf(md[i], d);
            md[i] = m;
            if (m > best) { best = m; bestn = i * 256 + tid; }  // strict > => lowest n per thread
        }

        // wave32 shuffle argmax (value desc, tie -> lower index), no barriers
#pragma unroll
        for (int off = 16; off > 0; off >>= 1) {
            float ov = __shfl_down(best, (unsigned)off, 32);
            int   oi = __shfl_down(bestn, (unsigned)off, 32);
            if (ov > best || (ov == best && oi < bestn)) { best = ov; bestn = oi; }
        }
        if (lane == 0) { sv[wave] = best; si[wave] = bestn; }
        __syncthreads();   // (2) sv/si visible

        // every thread reduces the 8 wave results -> identical `last` everywhere
        float fv = sv[0]; int fi = si[0];
#pragma unroll
        for (int w = 1; w < 8; ++w) {
            float ov = sv[w]; int oi = si[w];
            if (ov > fv || (ov == fv && oi < fi)) { fv = ov; fi = oi; }
        }
        last = fi;
        if (tid == 0) sel[step] = last;
        // next iteration's barrier (1) fences sel/sv/si before rewrites
    }
    __syncthreads();

    for (int s = tid; s < NPOINT; s += 256) {
        int id = sel[s];
        new_xyz[((size_t)b * NPOINT + s) * 3 + 0] = xb[id * 3 + 0];
        new_xyz[((size_t)b * NPOINT + s) * 3 + 1] = xb[id * 3 + 1];
        new_xyz[((size_t)b * NPOINT + s) * 3 + 2] = xb[id * 3 + 2];
    }
}

// ---------------------------------------------------------------------------
// Kernel 3: ball query. One wave32 per group; ordered compaction of the first
// 32 in-radius indices via ballot + prefix popcount, padded with the first hit.
// ---------------------------------------------------------------------------
__global__ void __launch_bounds__(256)
ballq_kernel(const float* __restrict__ xyz, const float* __restrict__ new_xyz,
             int* __restrict__ idx_out) {
    const int wave = threadIdx.x >> 5;
    const int lane = threadIdx.x & 31;
    const int g = blockIdx.x * 8 + wave;           // 0 .. 8191
    const int b = g >> 10;
    const float* xb = xyz + (size_t)b * NPTS * 3;

    const float cx = new_xyz[g * 3 + 0];
    const float cy = new_xyz[g * 3 + 1];
    const float cz = new_xyz[g * 3 + 2];

    int cnt = 0;
    int first = 0;
    bool haveFirst = false;

    for (int base = 0; base < NPTS && cnt < NSAMPLE; base += 32) {
        int n = base + lane;
        float dx = xb[n * 3 + 0] - cx;
        float dy = xb[n * 3 + 1] - cy;
        float dz = xb[n * 3 + 2] - cz;
        bool valid = (dx * dx + dy * dy + dz * dz) < R2;
        unsigned mask = (unsigned)__ballot(valid);
        if (!haveFirst && mask) { first = base + __builtin_ctz(mask); haveFirst = true; }
        int before = __popc(mask & ((1u << lane) - 1u));
        int pos = cnt + before;
        if (valid && pos < NSAMPLE) idx_out[(size_t)g * NSAMPLE + pos] = n;
        cnt += __popc(mask);
    }
    if (cnt > NSAMPLE) cnt = NSAMPLE;
    int pos = cnt + lane;                           // pad remainder with first hit
    if (pos < NSAMPLE) idx_out[(size_t)g * NSAMPLE + pos] = haveFirst ? first : 0;
}

// ---------------------------------------------------------------------------
// Kernel 4: fused gather + BN + ReLU + [32x160]x[160x256] f16 WMMA GEMM +
// max-over-samples + bias + transposed store. One WG (8 waves) per group.
// ---------------------------------------------------------------------------
__global__ void __launch_bounds__(256)
group_gemm_kernel(const float* __restrict__ xyz, const float* __restrict__ feat,
                  const float* __restrict__ bias,
                  const float* __restrict__ bn_gamma, const float* __restrict__ bn_beta,
                  const float* __restrict__ bn_mean,  const float* __restrict__ bn_var,
                  const float* __restrict__ new_xyz, const int* __restrict__ idx,
                  const _Float16* __restrict__ Wh, float* __restrict__ out_feat) {
    __shared__ _Float16 Ash[NSAMPLE * KPAD];   // 32 x 160 f16 = 10 KB
    __shared__ int nidx_s[NSAMPLE];

    const int g = blockIdx.x;                  // 0 .. 8191
    const int b = g >> 10;
    const int s = g & 1023;
    const int tid = threadIdx.x;

    if (tid < NSAMPLE) nidx_s[tid] = idx[(size_t)g * NSAMPLE + tid];
    __syncthreads();

    const float cx = new_xyz[g * 3 + 0];
    const float cy = new_xyz[g * 3 + 1];
    const float cz = new_xyz[g * 3 + 2];
    const float* xb = xyz  + (size_t)b * NPTS * 3;
    const float* fb = feat + (size_t)b * NPTS * CFEAT;

    // Stage A: gather + BN + ReLU -> f16, K zero-padded to 160
    for (int e = tid; e < NSAMPLE * KPAD; e += 256) {
        int sj = e / KPAD;
        int c  = e - sj * KPAD;
        float v = 0.0f;
        if (c < CIN) {
            int n = nidx_s[sj];
            if (n > NPTS - 1) n = NPTS - 1;    // mimic JAX OOB clamp
            float raw;
            if (c < 3) {
                float p = xb[n * 3 + c];
                raw = p - (c == 0 ? cx : (c == 1 ? cy : cz));
            } else {
                raw = fb[(size_t)n * CFEAT + (c - 3)];
            }
            float scale = bn_gamma[c] * rsqrtf(bn_var[c] + BN_EPS);
            v = (raw - bn_mean[c]) * scale + bn_beta[c];
            v = fmaxf(v, 0.0f);
        }
        Ash[e] = (_Float16)v;
    }
    __syncthreads();

    const int wave = tid >> 5;
    const int lane = tid & 31;
    const int hi   = lane >> 4;                // half-wave select (K offset)
    const int lm   = lane & 15;                // row (A) / column (B,D) id

#pragma unroll
    for (int t = 0; t < 2; ++t) {
        const int nt = wave * 2 + t;           // N-tile 0..15
        v8f acc0 = {};                         // samples 0..15
        v8f acc1 = {};                         // samples 16..31
        const _Float16* wrow = Wh + (size_t)(nt * 16 + lm) * KPAD;

#pragma unroll
        for (int ks = 0; ks < KPAD / 32; ++ks) {
            const int kbase = ks * 32;
            // B fragment: column n = W row; two 8-half K runs per lane (same K pattern as A)
            union { v16h v; v8h h[2]; } bf;
            bf.h[0] = *(const v8h*)(wrow + kbase + hi * 8);
            bf.h[1] = *(const v8h*)(wrow + kbase + 16 + hi * 8);
            // A fragments: per lane two contiguous 8-half runs (K pattern per ISA)
            union { v16h v; v8h h[2]; } a0, a1;
            const _Float16* ar0 = Ash + (size_t)lm * KPAD + kbase + hi * 8;
            const _Float16* ar1 = Ash + (size_t)(16 + lm) * KPAD + kbase + hi * 8;
            a0.h[0] = *(const v8h*)(ar0);
            a0.h[1] = *(const v8h*)(ar0 + 16);
            a1.h[0] = *(const v8h*)(ar1);
            a1.h[1] = *(const v8h*)(ar1 + 16);

            acc0 = __builtin_amdgcn_wmma_f32_16x16x32_f16(false, a0.v, false, bf.v,
                                                          (short)0, acc0, false, false);
            acc1 = __builtin_amdgcn_wmma_f32_16x16x32_f16(false, a1.v, false, bf.v,
                                                          (short)0, acc1, false, false);
        }

        // max over all 32 samples of column (nt*16+lm):
        // 8 rows in each acc locally, cross-half via shfl_xor(16)
        float m = -3.4e38f;
#pragma unroll
        for (int r = 0; r < 8; ++r) {
            m = fmaxf(m, acc0[r]);
            m = fmaxf(m, acc1[r]);
        }
        m = fmaxf(m, __shfl_xor(m, 16, 32));

        if (hi == 0) {
            int o = nt * 16 + lm;
            out_feat[((size_t)b * COUT + o) * NPOINT + s] = m + bias[o];
        }
    }
}

// ---------------------------------------------------------------------------
extern "C" void kernel_launch(void* const* d_in, const int* in_sizes, int n_in,
                              void* d_out, int out_size, void* d_ws, size_t ws_size,
                              hipStream_t stream) {
    const float* xyz      = (const float*)d_in[0];
    const float* feat     = (const float*)d_in[1];
    const float* W        = (const float*)d_in[2];
    const float* bias     = (const float*)d_in[3];
    const float* bn_gamma = (const float*)d_in[4];
    const float* bn_beta  = (const float*)d_in[5];
    const float* bn_mean  = (const float*)d_in[6];
    const float* bn_var   = (const float*)d_in[7];

    float* out      = (float*)d_out;
    float* new_xyz  = out;                               // 8*1024*3
    float* new_feat = out + (size_t)BATCH * NPOINT * 3;  // 8*256*1024

    char* ws = (char*)d_ws;
    _Float16* Wh  = (_Float16*)ws;                       // 256*160*2 = 81920 B
    int*      idx = (int*)(ws + 81920);                  // 8*1024*32*4 = 1 MB

    prep_w_kernel<<<(COUT * KPAD + 255) / 256, 256, 0, stream>>>(W, Wh);
    fps_kernel<<<BATCH, 256, 0, stream>>>(xyz, new_xyz);
    ballq_kernel<<<(BATCH * NPOINT) / 8, 256, 0, stream>>>(xyz, new_xyz, idx);
    group_gemm_kernel<<<BATCH * NPOINT, 256, 0, stream>>>(
        xyz, feat, bias, bn_gamma, bn_beta, bn_mean, bn_var,
        new_xyz, idx, Wh, new_feat);
}
